// myModel_31774168056193
// MI455X (gfx1250) — matrix operand order repaired and checked by reading, not compiled, not run on hardware
//
#include <hip/hip_runtime.h>
#include <hip/hip_bf16.h>
#include <math.h>

#define VOCAB  32000
#define DMODEL 1024
#define NTOK   4096            // B*S = 2*2048
#define NCHUNK (VOCAB / 64)    // 500 per-row softmax partial chunks

typedef __attribute__((ext_vector_type(16))) __bf16 v16bf;
typedef __attribute__((ext_vector_type(8)))  __bf16 v8bf;
typedef __attribute__((ext_vector_type(8)))  float  v8f;

__device__ __forceinline__ unsigned short f2bf_rne(float f) {
    unsigned u = __float_as_uint(f);
    u += 0x7FFFu + ((u >> 16) & 1u);          // round-to-nearest-even
    return (unsigned short)(u >> 16);
}

// ---------------------------------------------------------------------------
// K1: convert f32 rows (len 1024) -> bf16.  grid.x = #rows, 256 threads.
// Each thread converts one float4 (256*4 = 1024 elements per row).
// ---------------------------------------------------------------------------
__global__ void cvt_rows_bf16(const float* __restrict__ src,
                              unsigned short* __restrict__ dst) {
    size_t row = blockIdx.x;
    float4 v = ((const float4*)(src + row * DMODEL))[threadIdx.x];
    union { unsigned short h[4]; uint2 u; } p;
    p.h[0] = f2bf_rne(v.x); p.h[1] = f2bf_rne(v.y);
    p.h[2] = f2bf_rne(v.z); p.h[3] = f2bf_rne(v.w);
    *(uint2*)(dst + row * DMODEL + (size_t)threadIdx.x * 4) = p.u;
}

// ---------------------------------------------------------------------------
// K2: X = bf16(embed[inputs])  -- one block per token row.
// ---------------------------------------------------------------------------
__global__ void gather_x_bf16(const int* __restrict__ tokens,
                              const float* __restrict__ embed,
                              unsigned short* __restrict__ xb) {
    int t = blockIdx.x;
    size_t row = (size_t)tokens[t];
    float4 v = ((const float4*)(embed + row * DMODEL))[threadIdx.x];
    union { unsigned short h[4]; uint2 u; } p;
    p.h[0] = f2bf_rne(v.x); p.h[1] = f2bf_rne(v.y);
    p.h[2] = f2bf_rne(v.z); p.h[3] = f2bf_rne(v.w);
    *(uint2*)(xb + (size_t)t * DMODEL + (size_t)threadIdx.x * 4) = p.u;
}

// ---------------------------------------------------------------------------
// K3: logits = X(bf16) * W(bf16)^T + b   via v_wmma_f32_16x16x32_bf16,
//     fused partial softmax (per-row max / sum-exp over this wave's 64 cols).
//
// Block: 256 thr = 8 waves, arranged 4(M) x 2(N); each wave: 2 Msub x 4 Nsub
// of 16x16 -> block tile 128(M) x 128(N).  K loop: 1024 in steps of 32.
// ---------------------------------------------------------------------------
__global__ __launch_bounds__(256)
void lm_head_gemm(const unsigned short* __restrict__ xb,   // [NTOK][DMODEL]
                  const unsigned short* __restrict__ wb,   // [VOCAB][DMODEL]
                  const float* __restrict__ bias,          // [VOCAB]
                  float* __restrict__ logits,              // [NTOK][VOCAB]
                  float* __restrict__ pmax,                // [NTOK][NCHUNK]
                  float* __restrict__ psum) {              // [NTOK][NCHUNK]
    const int lane = threadIdx.x & 31;
    const int wave = threadIdx.x >> 5;
    const int wm   = wave & 3;                // 4 M-waves
    const int wn   = wave >> 2;               // 2 N-waves
    const int m_base = blockIdx.y * 128 + wm * 32;
    const int n_base = blockIdx.x * 128 + wn * 64;
    const int l16 = lane & 15;
    const int hi  = lane >> 4;                // half-wave select

    // A operand (16x32 bf16): lane<16 -> K 0..15,
    // lane>=16 -> K 16..31, row M = lane%16.
    const unsigned short* pa[2];
#pragma unroll
    for (int t = 0; t < 2; ++t)
        pa[t] = xb + (size_t)(m_base + t * 16 + l16) * DMODEL + hi * 16;

    // B operand (32x16 bf16): lane<16 -> K 0..15 of col N=lane,
    // lane>=16 -> K 16..31 of col N=lane-16 (32 contiguous bytes per lane).
    const unsigned short* pb[4];
#pragma unroll
    for (int s = 0; s < 4; ++s)
        pb[s] = wb + (size_t)(n_base + s * 16 + l16) * DMODEL + hi * 16;

    v8f acc[2][4] = {};

    for (int k0 = 0; k0 < DMODEL; k0 += 32) {
        v16bf a[2];
#pragma unroll
        for (int t = 0; t < 2; ++t) {
            v8bf lo = *(const v8bf*)(pa[t] + k0);
            v8bf hh = *(const v8bf*)(pa[t] + k0 + 8);
            a[t] = __builtin_shufflevector(lo, hh,
                    0,1,2,3,4,5,6,7,8,9,10,11,12,13,14,15);
        }
#pragma unroll
        for (int s = 0; s < 4; ++s) {
            v16bf bm = *(const v16bf*)(pb[s] + k0);
            __builtin_prefetch(pb[s] + k0 + 256, 0, 3);  // global_prefetch_b8
#pragma unroll
            for (int t = 0; t < 2; ++t)
                acc[t][s] = __builtin_amdgcn_wmma_f32_16x16x32_bf16(
                    false, a[t], false, bm, (short)0, acc[t][s], false, false);
        }
    }

    // Epilogue: +bias, store logits, partial softmax over this wave's 64 cols.
    float bv[4];
#pragma unroll
    for (int s = 0; s < 4; ++s) bv[s] = bias[n_base + s * 16 + l16];

    const int chunk = n_base >> 6;            // 64-col chunk index
#pragma unroll
    for (int t = 0; t < 2; ++t) {
#pragma unroll
        for (int i = 0; i < 8; ++i) {
            // C layout: VGPR i -> row M=i (lanes 0-15) / M=i+8 (lanes 16-31),
            // col N = lane%16.
            float v[4];
#pragma unroll
            for (int s = 0; s < 4; ++s) v[s] = acc[t][s][i] + bv[s];

            int row = m_base + t * 16 + i + hi * 8;
            size_t obase = (size_t)row * VOCAB + n_base + l16;
#pragma unroll
            for (int s = 0; s < 4; ++s) logits[obase + s * 16] = v[s];

            // Row reduction stays inside each 16-lane half (xor 1/2/4/8).
            float m = fmaxf(fmaxf(v[0], v[1]), fmaxf(v[2], v[3]));
#pragma unroll
            for (int off = 1; off < 16; off <<= 1)
                m = fmaxf(m, __shfl_xor(m, off, 32));
            float sum = 0.f;
#pragma unroll
            for (int s = 0; s < 4; ++s) sum += __expf(v[s] - m);
#pragma unroll
            for (int off = 1; off < 16; off <<= 1)
                sum += __shfl_xor(sum, off, 32);

            if (l16 == 0) {
                pmax[(size_t)row * NCHUNK + chunk] = m;
                psum[(size_t)row * NCHUNK + chunk] = sum;
            }
        }
    }
}

// ---------------------------------------------------------------------------
// K4: per-row logsumexp over 500 chunks + masked NLL accumulation.
// 128 threads = 4 waves; one wave per row.
// ---------------------------------------------------------------------------
__global__ void row_reduce(const float* __restrict__ pmax,
                           const float* __restrict__ psum,
                           const float* __restrict__ logits,
                           const int* __restrict__ labels,
                           const int* __restrict__ pad,
                           float* __restrict__ accum) {
    int lane = threadIdx.x & 31;
    int row  = blockIdx.x * 4 + (threadIdx.x >> 5);
    const float* pm = pmax + (size_t)row * NCHUNK;
    const float* ps = psum + (size_t)row * NCHUNK;

    float m = -INFINITY;
    for (int c = lane; c < NCHUNK; c += 32) m = fmaxf(m, pm[c]);
#pragma unroll
    for (int off = 1; off < 32; off <<= 1) m = fmaxf(m, __shfl_xor(m, off, 32));

    float s = 0.f;
    for (int c = lane; c < NCHUNK; c += 32) s += ps[c] * __expf(pm[c] - m);
#pragma unroll
    for (int off = 1; off < 32; off <<= 1) s += __shfl_xor(s, off, 32);

    if (lane == 0) {
        int lab = labels[row];
        if (lab != *pad) {
            float logZ = m + __logf(s);
            float ll = logits[(size_t)row * VOCAB + lab];
            atomicAdd(&accum[0], logZ - ll);   // nll
            atomicAdd(&accum[1], 1.0f);        // valid count
        }
    }
}

__global__ void finalize_loss(const float* __restrict__ accum,
                              float* __restrict__ out) {
    out[0] = accum[0] / fmaxf(accum[1], 1.0f);
}

// ---------------------------------------------------------------------------
extern "C" void kernel_launch(void* const* d_in, const int* in_sizes, int n_in,
                              void* d_out, int out_size, void* d_ws, size_t ws_size,
                              hipStream_t stream) {
    const int*   inputs = (const int*)  d_in[0];
    const int*   labels = (const int*)  d_in[1];
    const float* embed  = (const float*)d_in[2];
    const float* W      = (const float*)d_in[3];
    const float* bias   = (const float*)d_in[4];
    const int*   pad    = (const int*)  d_in[5];

    float* out    = (float*)d_out;
    float* logits = out + 1;                       // [NTOK*VOCAB], after loss

    char* ws = (char*)d_ws;
    unsigned short* wb   = (unsigned short*)ws;                              // 65,536,000 B
    unsigned short* xb   = (unsigned short*)(ws + 65536000);                 //  8,388,608 B
    float*          pmax = (float*)(ws + 65536000 + 8388608);                //  8,192,000 B
    float*          psum = (float*)(ws + 65536000 + 8388608 + 8192000);      //  8,192,000 B
    float*          acc  = (float*)(ws + 65536000 + 8388608 + 2 * 8192000);  //         8 B

    cvt_rows_bf16<<<VOCAB, 256, 0, stream>>>(W, wb);
    gather_x_bf16<<<NTOK, 256, 0, stream>>>(inputs, embed, xb);
    hipMemsetAsync(acc, 0, 2 * sizeof(float), stream);

    dim3 grid(VOCAB / 128, NTOK / 128);            // 250 x 32
    lm_head_gemm<<<grid, 256, 0, stream>>>(xb, wb, bias, logits, pmax, psum);

    row_reduce<<<NTOK / 4, 128, 0, stream>>>(pmax, psum, logits, labels, pad, acc);
    finalize_loss<<<1, 1, 0, stream>>>(acc, out);
}
